// window_attention_8254927143429
// MI455X (gfx1250) — matrix operand order repaired and checked by reading, not compiled, not run on hardware
//
#include <hip/hip_runtime.h>
#include <hip/hip_bf16.h>

// ---------------------------------------------------------------------------
// 3D window attention (Swin-style), fully fused into 4 kernels, all matmuls
// via v_wmma_f32_16x16x32_f16 (CDNA5 / gfx1250, wave32).
//
// Shapes: B=512 windows, N=343 tokens, C=192, H=6 heads, hd=32, NW=64 masks.
// Workspace usage (~273 MB): Q,K,V,O as f16 + per-head bias planes f32.
// ---------------------------------------------------------------------------

#define N_TOK 343
#define NPAD  352          // 22 * 16
#define HD    32
#define NH    6
#define CDIM  192
#define NN    (N_TOK * N_TOK)   // 117649
#define QKV_SCALE 0.17677669529663687f  // 1/sqrt(32)

typedef _Float16 h8   __attribute__((ext_vector_type(8)));
typedef _Float16 v16h __attribute__((ext_vector_type(16)));
typedef float    v8f  __attribute__((ext_vector_type(8)));
typedef float    f4   __attribute__((ext_vector_type(4)));

static __device__ __forceinline__ v16h mk16(h8 lo, h8 hh) {
    v16h r;
#pragma unroll
    for (int i = 0; i < 8; ++i) { r[i] = lo[i]; r[i + 8] = hh[i]; }
    return r;
}

static __device__ __forceinline__ v8f wmma16(v16h a, v16h b, v8f c) {
    // D = A(16x32 f16) * B(32x16 f16) + C(16x16 f32)
    return __builtin_amdgcn_wmma_f32_16x16x32_f16(
        /*neg_a=*/false, a, /*neg_b=*/false, b,
        /*c_mod=*/(short)0, c, /*reuse_a=*/false, /*reuse_b=*/false);
}

// ---------------------------------------------------------------------------
// Kernel 0: expand relative-position bias into per-head (H,N,N) f32 planes.
// ---------------------------------------------------------------------------
__global__ __launch_bounds__(256) void bias_expand_kernel(
    const float* __restrict__ tbl, const int* __restrict__ rpi,
    float* __restrict__ biasH) {
    int i = blockIdx.x * 256 + threadIdx.x;
    if (i < NN) {
        int r = rpi[i];
#pragma unroll
        for (int h = 0; h < NH; ++h)
            biasH[h * NN + i] = tbl[r * NH + h];
    }
}

// ---------------------------------------------------------------------------
// Kernel 1: QKV GEMM  X(175616x192) @ W(192x576) + b  ->  Q,K,V f16
// Output layout: (B, H, N, hd), Q pre-scaled by 1/sqrt(hd).
// Grid: (2744, 6 strips of 96 cols), block 256 (8 waves).
// Wave w: mtile = bx*4 + w/2, handles 3 N-tiles (48 cols) of its strip.
// ---------------------------------------------------------------------------
__global__ __launch_bounds__(256) void qkv_kernel(
    const float* __restrict__ x, const float* __restrict__ wq,
    const float* __restrict__ bq,
    _Float16* __restrict__ qb, _Float16* __restrict__ kb,
    _Float16* __restrict__ vb) {
    __shared__ __align__(16) _Float16 Wt[96 * 192];  // [n_local][k], 36 KB
    const int strip = blockIdx.y;
    const int tid = threadIdx.x;
    for (int i = tid; i < 96 * 192; i += 256) {
        int k = i / 96, n = i % 96;                  // coalesced global read
        Wt[n * 192 + k] = (_Float16)wq[k * 576 + strip * 96 + n];
    }
    __syncthreads();

    const int wave = tid >> 5, lane = tid & 31;
    const int n16 = lane & 15, hi = lane >> 4;
    const int mtile = blockIdx.x * 4 + (wave >> 1);
    const int ngrp = wave & 1;
    const int row = mtile * 16 + n16;                // A-matrix row (=lane)
    const float* xr = x + (size_t)row * CDIM;

    v8f acc[3];
#pragma unroll
    for (int nl = 0; nl < 3; ++nl)
#pragma unroll
        for (int g = 0; g < 8; ++g) acc[nl][g] = 0.f;

#pragma unroll
    for (int kk = 0; kk < 6; ++kk) {  // K = 192 = 6 * 32
        // A fragment: ISA layout K = e + (e<8?0:8) + 8*hi
        f4 a00 = *(const f4*)(xr + kk * 32 + 8 * hi);
        f4 a01 = *(const f4*)(xr + kk * 32 + 8 * hi + 4);
        f4 a10 = *(const f4*)(xr + kk * 32 + 16 + 8 * hi);
        f4 a11 = *(const f4*)(xr + kk * 32 + 20 + 8 * hi);
        v16h A;
#pragma unroll
        for (int i = 0; i < 4; ++i) {
            A[i]      = (_Float16)a00[i];
            A[i + 4]  = (_Float16)a01[i];
            A[i + 8]  = (_Float16)a10[i];
            A[i + 12] = (_Float16)a11[i];
        }
#pragma unroll
        for (int nl = 0; nl < 3; ++nl) {
            int ncol = (ngrp * 3 + nl) * 16 + n16;   // B col (=lane), K = e + (e<8?0:8) + 8*hi
            h8 w0 = *(const h8*)(Wt + ncol * 192 + kk * 32 + 8 * hi);
            h8 w1 = *(const h8*)(Wt + ncol * 192 + kk * 32 + 16 + 8 * hi);
            acc[nl] = wmma16(A, mk16(w0, w1), acc[nl]);
        }
    }

#pragma unroll
    for (int nl = 0; nl < 3; ++nl) {
        int clo = strip * 96 + (ngrp * 3 + nl) * 16;  // 16-aligned global col
        int s = clo / 192;                            // 0=Q,1=K,2=V (uniform)
        int rem = clo - s * 192;
        int hh = rem >> 5;
        int dlo = rem & 31;
        _Float16* dst = (s == 0) ? qb : (s == 1 ? kb : vb);
#pragma unroll
        for (int g = 0; g < 8; ++g) {  // C layout: M = g + 8*hi, Ncol = n16
            int R = mtile * 16 + g + 8 * hi;
            int bb = R / N_TOK, tok = R - bb * N_TOK;
            float v = acc[nl][g] + bq[clo + n16];
            if (s == 0) v *= QKV_SCALE;
            dst[((size_t)(bb * NH + hh) * N_TOK + tok) * HD + dlo + n16] =
                (_Float16)v;
        }
    }
}

// ---------------------------------------------------------------------------
// Kernel 2: attention. One workgroup per (b,h); 4 waves, each wave owns
// 16-row query tiles (stride 4). Scores kept register-resident (22 x v8f),
// softmax via shfl-xor within 16-lane groups, P staged f16 in LDS for P@V.
// ---------------------------------------------------------------------------
__global__ __launch_bounds__(128) void attn_kernel(
    const _Float16* __restrict__ qb, const _Float16* __restrict__ kb,
    const _Float16* __restrict__ vb, const float* __restrict__ biasH,
    const float* __restrict__ mask, _Float16* __restrict__ ob) {
    __shared__ __align__(16) _Float16 pbuf[4][16 * NPAD];  // 45 KB
    const int bh = blockIdx.x;
    const int b = bh / NH, h = bh - b * NH;
    const int w = b & 63;  // B = 8 * NW, window id = b % 64
    const int tid = threadIdx.x;
    const int wave = tid >> 5, lane = tid & 31;
    const int n16 = lane & 15, hi = lane >> 4;

    const _Float16* Qb = qb + (size_t)bh * (N_TOK * HD);
    const _Float16* Kb = kb + (size_t)bh * (N_TOK * HD);
    const _Float16* Vb = vb + (size_t)bh * (N_TOK * HD);
    const float* Bh = biasH + (size_t)h * NN;
    const float* Mw = mask + (size_t)w * NN;
    _Float16* P = pbuf[wave];

    for (int mt = wave; mt < 22; mt += 4) {
        __builtin_prefetch(Mw + (mt * 16 < N_TOK ? mt * 16 : 342) * N_TOK, 0, 0);

        // ---- S = Q * K^T  (hd = 32 -> a single WMMA K-step per tile)
        int ra = mt * 16 + n16; ra = ra > 342 ? 342 : ra;  // clamp pad rows
        h8 q0 = *(const h8*)(Qb + ra * HD + 8 * hi);
        h8 q1 = *(const h8*)(Qb + ra * HD + 16 + 8 * hi);
        v16h Aq = mk16(q0, q1);

        v8f acc[22];
#pragma unroll
        for (int t = 0; t < 22; ++t) {
            int rb = t * 16 + n16; rb = rb > 342 ? 342 : rb;
            h8 k0 = *(const h8*)(Kb + rb * HD + 8 * hi);
            h8 k1 = *(const h8*)(Kb + rb * HD + 16 + 8 * hi);
            v8f z; 
#pragma unroll
            for (int g = 0; g < 8; ++g) z[g] = 0.f;
            acc[t] = wmma16(Aq, mk16(k0, k1), z);
        }

        // ---- +bias +mask, row softmax. Row m = g + 8*hi (C layout).
#pragma unroll
        for (int g = 0; g < 8; ++g) {
            int rq = mt * 16 + g + 8 * hi; rq = rq > 342 ? 342 : rq;
            const float* br = Bh + rq * N_TOK;
            const float* mr = Mw + rq * N_TOK;
            float mx = -3.0e38f;
#pragma unroll
            for (int t = 0; t < 22; ++t) {
                int col = t * 16 + n16;
                float v;
                if (col < N_TOK) v = acc[t][g] + br[col] + mr[col];
                else             v = -__builtin_inff();   // pad cols -> exp 0
                acc[t][g] = v;
                mx = fmaxf(mx, v);
            }
            mx = fmaxf(mx, __shfl_xor(mx, 1, 32));
            mx = fmaxf(mx, __shfl_xor(mx, 2, 32));
            mx = fmaxf(mx, __shfl_xor(mx, 4, 32));
            mx = fmaxf(mx, __shfl_xor(mx, 8, 32));
            float s = 0.f;
#pragma unroll
            for (int t = 0; t < 22; ++t) {
                float e = __expf(acc[t][g] - mx);
                acc[t][g] = e;
                s += e;
            }
            s += __shfl_xor(s, 1, 32);
            s += __shfl_xor(s, 2, 32);
            s += __shfl_xor(s, 4, 32);
            s += __shfl_xor(s, 8, 32);
            float is = 1.0f / s;
            int m = g + 8 * hi;
#pragma unroll
            for (int t = 0; t < 22; ++t)
                P[m * NPAD + t * 16 + n16] = (_Float16)(acc[t][g] * is);
        }
        // per-wave LDS buffer, same-wave RAW: LDS ops in-order, no barrier.

        // ---- O = P (16x352) * V (352x32)
#pragma unroll
        for (int nt = 0; nt < 2; ++nt) {
            int d0 = nt * 16 + n16;
            v8f oacc;
#pragma unroll
            for (int g = 0; g < 8; ++g) oacc[g] = 0.f;
#pragma unroll
            for (int ks = 0; ks < 11; ++ks) {
                h8 p0 = *(const h8*)(P + n16 * NPAD + ks * 32 + 8 * hi);
                h8 p1 = *(const h8*)(P + n16 * NPAD + ks * 32 + 16 + 8 * hi);
                v16h Ap = mk16(p0, p1);
                v16h Bv;
#pragma unroll
                for (int e = 0; e < 16; ++e) {  // B: col=d0, K = e + (e<8?0:8) + 8*hi
                    int j = ks * 32 + 8 * hi + e + (e < 8 ? 0 : 8);
                    j = j > 342 ? 342 : j;      // pad keys: P==0 there
                    Bv[e] = Vb[j * HD + d0];
                }
                oacc = wmma16(Ap, Bv, oacc);
            }
#pragma unroll
            for (int g = 0; g < 8; ++g) {
                int tok = mt * 16 + g + 8 * hi;
                if (tok < N_TOK)
                    ob[((size_t)b * N_TOK + tok) * CDIM + h * HD + d0] =
                        (_Float16)oacc[g];
            }
        }
    }
}

// ---------------------------------------------------------------------------
// Kernel 3: output projection O(175616x192,f16) @ proj_w(192x192) + b -> f32
// Grid: (2744, 2 strips of 96 cols), block 256.
// ---------------------------------------------------------------------------
__global__ __launch_bounds__(256) void proj_kernel(
    const _Float16* __restrict__ ob, const float* __restrict__ pw,
    const float* __restrict__ pb, float* __restrict__ out) {
    __shared__ __align__(16) _Float16 Wt[96 * 192];
    const int strip = blockIdx.y;
    const int tid = threadIdx.x;
    for (int i = tid; i < 96 * 192; i += 256) {
        int k = i / 96, n = i % 96;
        Wt[n * 192 + k] = (_Float16)pw[k * 192 + strip * 96 + n];
    }
    __syncthreads();

    const int wave = tid >> 5, lane = tid & 31;
    const int n16 = lane & 15, hi = lane >> 4;
    const int mtile = blockIdx.x * 4 + (wave >> 1);
    const int ngrp = wave & 1;
    const int row = mtile * 16 + n16;
    const _Float16* orow = ob + (size_t)row * CDIM;

    v8f acc[3];
#pragma unroll
    for (int nl = 0; nl < 3; ++nl)
#pragma unroll
        for (int g = 0; g < 8; ++g) acc[nl][g] = 0.f;

#pragma unroll
    for (int kk = 0; kk < 6; ++kk) {
        h8 a0 = *(const h8*)(orow + kk * 32 + 8 * hi);
        h8 a1 = *(const h8*)(orow + kk * 32 + 16 + 8 * hi);
        v16h A = mk16(a0, a1);
#pragma unroll
        for (int nl = 0; nl < 3; ++nl) {
            int ncol = (ngrp * 3 + nl) * 16 + n16;
            h8 w0 = *(const h8*)(Wt + ncol * 192 + kk * 32 + 8 * hi);
            h8 w1 = *(const h8*)(Wt + ncol * 192 + kk * 32 + 16 + 8 * hi);
            acc[nl] = wmma16(A, mk16(w0, w1), acc[nl]);
        }
    }

#pragma unroll
    for (int nl = 0; nl < 3; ++nl) {
        int c = strip * 96 + (ngrp * 3 + nl) * 16 + n16;
#pragma unroll
        for (int g = 0; g < 8; ++g) {
            int R = mtile * 16 + g + 8 * hi;
            out[(size_t)R * CDIM + c] = acc[nl][g] + pb[c];
        }
    }
}

// ---------------------------------------------------------------------------
// Launcher. Inputs (setup_inputs order):
//  0:x (B,N,C) f32   1:mask (64,N,N) f32   2:qkv_w (192,576)  3:qkv_b (576)
//  4:proj_w (192,192) 5:proj_b (192)  6:bias_table (2197,6)  7:rpi (N,N) i32
// Workspace layout (needs ~273 MB): Q | K | V | O (f16) | biasH (f32)
// ---------------------------------------------------------------------------
extern "C" void kernel_launch(void* const* d_in, const int* in_sizes, int n_in,
                              void* d_out, int out_size, void* d_ws,
                              size_t ws_size, hipStream_t stream) {
    (void)in_sizes; (void)n_in; (void)out_size; (void)ws_size;
    const float* x          = (const float*)d_in[0];
    const float* mask       = (const float*)d_in[1];
    const float* qkv_w      = (const float*)d_in[2];
    const float* qkv_b      = (const float*)d_in[3];
    const float* proj_w     = (const float*)d_in[4];
    const float* proj_b     = (const float*)d_in[5];
    const float* bias_table = (const float*)d_in[6];
    const int*   rpi        = (const int*)d_in[7];
    float* out = (float*)d_out;

    const size_t QSZ = (size_t)512 * NH * N_TOK * HD;  // 33,718,272 halfs
    _Float16* qb = (_Float16*)d_ws;
    _Float16* kb = qb + QSZ;
    _Float16* vb = kb + QSZ;
    _Float16* ob = vb + QSZ;       // (B*N, 192) f16 — same element count
    float* biasH = (float*)(ob + QSZ);

    bias_expand_kernel<<<(NN + 255) / 256, 256, 0, stream>>>(bias_table, rpi,
                                                             biasH);
    qkv_kernel<<<dim3(2744, 6), 256, 0, stream>>>(x, qkv_w, qkv_b, qb, kb, vb);
    attn_kernel<<<dim3(512 * NH), 128, 0, stream>>>(qb, kb, vb, biasH, mask,
                                                    ob);
    proj_kernel<<<dim3(2744, 2), 256, 0, stream>>>(ob, proj_w, proj_b, out);
}
